// MultiHead_RNN_16329465660026
// MI455X (gfx1250) — matrix operand order repaired and checked by reading, not compiled, not run on hardware
//
#include <hip/hip_runtime.h>
#include <hip/hip_bf16.h>

// ---------------------------------------------------------------------------
// MultiHead RNN, MI455X (gfx1250) wave32 WMMA bf16 pipeline
//   E = relu(x @ proj_w^T + proj_b)  -> h0 (2048 x 128, rows = h*256+b)
//   4x: h = relu(h @ W)   (step1 K=128, step4 N=896 -> feat layout)
//   out = feat @ out_w^T + out_b     (split-K + f32 atomics)
// TN bf16 GEMMs, fp32 accumulation (v_wmma_f32_16x16x32_bf16).
// Tile staging: gfx1250 GLOBAL_LOAD_ASYNC_TO_LDS_B128, 2-deep ping-pong
// pipeline unrolled by 2 so buffer indices are compile-time constants.
// All kChunk values used are multiples of 64 (even tile count).
// ---------------------------------------------------------------------------

typedef __attribute__((ext_vector_type(16))) __bf16 v16bf;
typedef __attribute__((ext_vector_type(8)))  float  v8f;

#define LDS_STRIDE 40  // 32 K-elems + 8 pad (bf16), 80 bytes/row (16B aligned)

// 16B global -> LDS async copy (per-lane LDS byte address, ASYNCcnt-tracked)
__device__ __forceinline__ void async_copy16(unsigned int lds, const void* g) {
  asm volatile("global_load_async_to_lds_b128 %0, %1, off"
               :: "v"(lds), "v"(g) : "memory");
}
__device__ __forceinline__ void async_wait4() {  // ASYNCcnt <= 4
  asm volatile("s_wait_asynccnt 4" ::: "memory");
}
__device__ __forceinline__ void async_wait0() {  // ASYNCcnt == 0
  asm volatile("s_wait_asynccnt 0" ::: "memory");
}

// A fragment (16x32 bf16, MxK): lane L(<16) = row M=L, K {0..7,16..23};
// lane L+16 same row, K {8..15,24..31}.
__device__ __forceinline__ v16bf frag_a(const __bf16* s, int row, int lane) {
  union { v16bf v; uint4 q[2]; } u;
  const uint4* p = (const uint4*)(s + row * LDS_STRIDE + ((lane & 16) ? 8 : 0));
  u.q[0] = p[0];
  u.q[1] = p[2];
  return u.v;
}
// B fragment (32x16 bf16, KxN): lane L(<16) = col N=L, K {0..7,16..23};
// lane L+16 same col, K {8..15,24..31}.
__device__ __forceinline__ v16bf frag_b(const __bf16* s, int row, int lane) {
  union { v16bf v; uint4 q[2]; } u;
  const uint4* p = (const uint4*)(s + row * LDS_STRIDE + ((lane & 16) ? 8 : 0));
  u.q[0] = p[0];
  u.q[1] = p[2];
  return u.v;
}

// stage one 128x32 tile pair chunk set (4 async 16B ops) and advance pointers
__device__ __forceinline__ void issue_tile(
    unsigned int ldsA, unsigned int ldsB, unsigned int halfStep,
    const __bf16*& aP0, const __bf16*& aP1,
    const __bf16*& bP0, const __bf16*& bP1) {
  async_copy16(ldsA, aP0);
  async_copy16(ldsA + halfStep, aP1);
  async_copy16(ldsB, bP0);
  async_copy16(ldsB + halfStep, bP1);
  aP0 += 32; aP1 += 32; bP0 += 32; bP1 += 32;
}

// 12 ds_load_b128 + 8 WMMA on one staged 128x32 tile pair
__device__ __forceinline__ void compute_tile(
    const __bf16* a, const __bf16* b, v8f (&acc)[4][2],
    int wm, int wn, int lane) {
  v16bf af[4], bfr[2];
#pragma unroll
  for (int i = 0; i < 4; ++i)
    af[i] = frag_a(a, wm * 64 + i * 16 + (lane & 15), lane);
#pragma unroll
  for (int j = 0; j < 2; ++j)
    bfr[j] = frag_b(b, wn * 32 + j * 16 + (lane & 15), lane);
#pragma unroll
  for (int i = 0; i < 4; ++i)
#pragma unroll
    for (int j = 0; j < 2; ++j)
      acc[i][j] = __builtin_amdgcn_wmma_f32_16x16x32_bf16(
          false, af[i], false, bfr[j], (short)0, acc[i][j], false, false);
}

// Epilogue modes
#define EPI_RELU 0   // outB[row*ldc+col] = bf16(relu(acc))
#define EPI_PROJ 1   // col=h*128+s: outB[(h*256+row)*128+s] = bf16(relu(acc+bias[col]))
#define EPI_FEAT 2   // row=h*256+b: outB[b*7168 + h*896 + col] = bf16(relu(acc))
#define EPI_OUT  3   // atomicAdd(outF[row*128+col], acc)   (split-K partials)

template <int EPI>
__global__ __launch_bounds__(256) void gemm_bf16_wmma(
    const __bf16* __restrict__ A, int lda,
    const __bf16* __restrict__ B, int ldb,   // B is N-major: B[n*ldb + k]
    int kChunk,                              // multiple of 64
    __bf16* __restrict__ outB, int ldc,
    const float* __restrict__ bias,
    float* __restrict__ outF) {
  __shared__ __align__(16) __bf16 sA[2][128 * LDS_STRIDE];
  __shared__ __align__(16) __bf16 sB[2][128 * LDS_STRIDE];

  const int t    = threadIdx.x;
  const int lane = t & 31;
  const int w    = t >> 5;       // 8 waves
  const int wm   = w & 1;        // 2 waves along M  (64 rows each)
  const int wn   = w >> 1;       // 4 waves along N  (32 cols each)
  const int m0   = blockIdx.y * 128;
  const int n0   = blockIdx.x * 128;
  const size_t kBase = (size_t)blockIdx.z * (size_t)kChunk;

  // per-thread staging: thread t moves 16B chunks for rows (t>>2), (t>>2)+64,
  // chunk column (t&3), of both A and B tiles.
  const int srow = t >> 2;       // 0..63
  const int scc  = t & 3;        // 16B chunk within the 64B (32-elem) K row
  const __bf16* aP0 = A + (size_t)(m0 + srow) * (size_t)lda + kBase + scc * 8;
  const __bf16* aP1 = aP0 + (size_t)64 * (size_t)lda;
  const __bf16* bP0 = B + (size_t)(n0 + srow) * (size_t)ldb + kBase + scc * 8;
  const __bf16* bP1 = bP0 + (size_t)64 * (size_t)ldb;
  const unsigned int stageOff =
      (unsigned int)srow * (LDS_STRIDE * 2) + (unsigned int)scc * 16;
  // generic shared pointer: addr[31:0] is the LDS byte offset (ISA aperture)
  const unsigned int ldsA0 = (unsigned int)(uintptr_t)(&sA[0][0]) + stageOff;
  const unsigned int ldsA1 = (unsigned int)(uintptr_t)(&sA[1][0]) + stageOff;
  const unsigned int ldsB0 = (unsigned int)(uintptr_t)(&sB[0][0]) + stageOff;
  const unsigned int ldsB1 = (unsigned int)(uintptr_t)(&sB[1][0]) + stageOff;
  const unsigned int halfStep = 64u * (LDS_STRIDE * 2);  // rows +64 in LDS

  v8f acc[4][2];
#pragma unroll
  for (int i = 0; i < 4; ++i)
#pragma unroll
    for (int j = 0; j < 2; ++j) acc[i][j] = v8f{};

  const int nPairs = kChunk >> 6;  // two 32-wide tiles per iteration

  // prologue: tile 0 -> buf0
  issue_tile(ldsA0, ldsB0, halfStep, aP0, aP1, bP0, bP1);

  for (int p = 0; p < nPairs; ++p) {
    // tile 2p+1 -> buf1 (always exists: even tile count)
    issue_tile(ldsA1, ldsB1, halfStep, aP0, aP1, bP0, bP1);
    async_wait4();                 // tile 2p resident in buf0
    __syncthreads();
    compute_tile(sA[0], sB[0], acc, wm, wn, lane);
    __syncthreads();               // all waves done reading buf0

    if (p + 1 < nPairs) {
      issue_tile(ldsA0, ldsB0, halfStep, aP0, aP1, bP0, bP1);  // tile 2p+2
      async_wait4();               // tile 2p+1 resident in buf1
    } else {
      async_wait0();
    }
    __syncthreads();
    compute_tile(sA[1], sB[1], acc, wm, wn, lane);
    __syncthreads();               // all waves done reading buf1
  }

  // --- epilogue: C layout = VGPR r -> M = r + (lane>=16 ? 8 : 0), N = lane&15
  const int rhi  = (lane & 16) ? 8 : 0;
  const int coll = lane & 15;
#pragma unroll
  for (int i = 0; i < 4; ++i) {
#pragma unroll
    for (int j = 0; j < 2; ++j) {
#pragma unroll
      for (int r = 0; r < 8; ++r) {
        const int row = m0 + wm * 64 + i * 16 + r + rhi;
        const int col = n0 + wn * 32 + j * 16 + coll;
        float v = acc[i][j][r];
        if (EPI == EPI_RELU) {
          v = v > 0.f ? v : 0.f;
          outB[(size_t)row * (size_t)ldc + col] = (__bf16)v;
        } else if (EPI == EPI_PROJ) {
          v += bias[col];
          v = v > 0.f ? v : 0.f;
          const int h = col >> 7, s = col & 127;
          outB[((size_t)h * 256 + row) * 128 + s] = (__bf16)v;
        } else if (EPI == EPI_FEAT) {
          v = v > 0.f ? v : 0.f;
          const int b = row & 255, h = row >> 8;
          outB[(size_t)b * 7168 + (size_t)h * 896 + col] = (__bf16)v;
        } else {  // EPI_OUT
          atomicAdd(outF + (size_t)row * 128 + col, v);
        }
      }
    }
  }
}

// W (4096x4096 f32, row-major d,e) -> Wt (bf16, row-major e,d) via 32x32 LDS tile
__global__ __launch_bounds__(256) void transpose_convert_w(
    const float* __restrict__ W, __bf16* __restrict__ Wt) {
  __shared__ float tile[32][33];
  const int tx = threadIdx.x & 31;
  const int ty = threadIdx.x >> 5;  // 0..7
#pragma unroll
  for (int i = 0; i < 4; ++i) {
    const int d = blockIdx.y * 32 + ty + i * 8;
    const int e = blockIdx.x * 32 + tx;
    tile[ty + i * 8][tx] = W[(size_t)d * 4096 + e];
  }
  __syncthreads();
#pragma unroll
  for (int i = 0; i < 4; ++i) {
    const int e = blockIdx.x * 32 + ty + i * 8;
    const int d = blockIdx.y * 32 + tx;
    Wt[(size_t)e * 4096 + d] = (__bf16)tile[tx][ty + i * 8];
  }
}

__global__ __launch_bounds__(256) void convert_f32_bf16(
    const float* __restrict__ in, __bf16* __restrict__ out, int n) {
  const int i = blockIdx.x * 256 + threadIdx.x;
  if (i < n) out[i] = (__bf16)in[i];
}

__global__ __launch_bounds__(256) void init_out_bias(
    float* __restrict__ out, const float* __restrict__ bias) {
  const int i = blockIdx.x * 256 + threadIdx.x;  // 256*128
  out[i] = bias[i & 127];
}

// ---------------------------------------------------------------------------
extern "C" void kernel_launch(void* const* d_in, const int* in_sizes, int n_in,
                              void* d_out, int out_size, void* d_ws, size_t ws_size,
                              hipStream_t stream) {
  const float* x      = (const float*)d_in[0];  // 256x768
  const float* W      = (const float*)d_in[1];  // 4096x4096
  const float* proj_w = (const float*)d_in[2];  // 8x128x768
  const float* proj_b = (const float*)d_in[3];  // 8x128 (=1024)
  const float* out_w  = (const float*)d_in[4];  // 128x7168
  const float* out_b  = (const float*)d_in[5];  // 128
  float* out          = (float*)d_out;          // 256x128

  // workspace layout (bytes)
  constexpr size_t OFF_WT   = 0;                            // 4096*4096*2 = 33554432
  constexpr size_t OFF_HA   = 33554432;                     // 2048*4096*2 = 16777216
  constexpr size_t OFF_HB   = OFF_HA + 16777216;            // 16777216
  constexpr size_t OFF_H0   = OFF_HB + 16777216;            // 2048*128*2 = 524288
  constexpr size_t OFF_XB   = OFF_H0 + 524288;              // 256*768*2 = 393216
  constexpr size_t OFF_PW   = OFF_XB + 393216;              // 1024*768*2 = 1572864
  constexpr size_t OFF_OW   = OFF_PW + 1572864;             // 128*7168*2 = 1835008
  constexpr size_t OFF_FEAT = OFF_OW + 1835008;             // 256*7168*2 = 3670016
  constexpr size_t WS_NEED  = OFF_FEAT + 3670016;           // ~71.6 MB
  if (ws_size < WS_NEED) return;

  char* ws = (char*)d_ws;
  __bf16* Wt   = (__bf16*)(ws + OFF_WT);
  __bf16* hA   = (__bf16*)(ws + OFF_HA);
  __bf16* hB   = (__bf16*)(ws + OFF_HB);
  __bf16* h0   = (__bf16*)(ws + OFF_H0);
  __bf16* xb   = (__bf16*)(ws + OFF_XB);
  __bf16* pw   = (__bf16*)(ws + OFF_PW);
  __bf16* ow   = (__bf16*)(ws + OFF_OW);
  __bf16* feat = (__bf16*)(ws + OFF_FEAT);

  // 1) precision / layout conversion
  transpose_convert_w<<<dim3(128, 128), 256, 0, stream>>>(W, Wt);
  convert_f32_bf16<<<(196608 + 255) / 256, 256, 0, stream>>>(x, xb, 196608);
  convert_f32_bf16<<<(786432 + 255) / 256, 256, 0, stream>>>(proj_w, pw, 786432);
  convert_f32_bf16<<<(917504 + 255) / 256, 256, 0, stream>>>(out_w, ow, 917504);
  init_out_bias<<<128, 256, 0, stream>>>(out, out_b);

  // 2) projection: (256 x 1024) = x(256x768) @ proj_w^T ; bias+relu -> h0
  gemm_bf16_wmma<EPI_PROJ><<<dim3(8, 2, 1), 256, 0, stream>>>(
      xb, 768, pw, 768, 768, h0, 128, proj_b, nullptr);

  // 3) recurrence (rows = h*256+b)
  // step1: K=128 (only first 128 cols of initial h are nonzero)
  gemm_bf16_wmma<EPI_RELU><<<dim3(32, 16, 1), 256, 0, stream>>>(
      h0, 128, Wt, 4096, 128, hA, 4096, nullptr, nullptr);
  // steps 2,3: full 2048 x 4096 x 4096
  gemm_bf16_wmma<EPI_RELU><<<dim3(32, 16, 1), 256, 0, stream>>>(
      hA, 4096, Wt, 4096, 4096, hB, 4096, nullptr, nullptr);
  gemm_bf16_wmma<EPI_RELU><<<dim3(32, 16, 1), 256, 0, stream>>>(
      hB, 4096, Wt, 4096, 4096, hA, 4096, nullptr, nullptr);
  // step4: only output cols 3200..4095 (N=896) -> feat[b, h*896+j]
  gemm_bf16_wmma<EPI_FEAT><<<dim3(7, 16, 1), 256, 0, stream>>>(
      hA, 4096, Wt + (size_t)3200 * 4096, 4096, 4096, feat, 0, nullptr, nullptr);

  // 4) head: out(256x128) = feat(256x7168) @ out_w^T + out_b, split-K=14
  gemm_bf16_wmma<EPI_OUT><<<dim3(1, 2, 14), 256, 0, stream>>>(
      feat, 7168, ow, 7168, 512, nullptr, 0, nullptr, out);
}
